// MultiHeadLatentAttention_50182397886789
// MI455X (gfx1250) — matrix operand order repaired and checked by reading, not compiled, not run on hardware
//
#include <hip/hip_runtime.h>
#include <hip/hip_bf16.h>
#include <math.h>

// ---------------------------------------------------------------------------
// MLA forward for MI455X (gfx1250):
//  - all big contractions via v_wmma_f32_16x16x32_bf16 (f32 accumulate)
//  - straight-copy tiles staged with global_load_async_to_lds_b64 (ASYNCcnt)
//  - transposed tiles staged via register-batched, software-pipelined stores
// ---------------------------------------------------------------------------

typedef __attribute__((ext_vector_type(16))) __bf16 v16bf;
typedef __attribute__((ext_vector_type(8)))  float  v8f;

#define S_LEN 4096
#define HIDN  2048
#define NH    16
#define HD    128

// ---- CDNA5 async global->LDS copy (tracked by ASYNCcnt) --------------------

__device__ __forceinline__ void async_b64(void* lds, const void* gptr) {
  // VDST = per-lane LDS byte offset (low 32 bits of the flat shared address),
  // VADDR = 64-bit global address.  ISA 15.18.3 opcode 97.
  asm volatile("global_load_async_to_lds_b64 %0, %1, off"
               :: "v"((unsigned)(unsigned long long)(uintptr_t)lds),
                  "v"((unsigned long long)(uintptr_t)gptr)
               : "memory");
}

__device__ __forceinline__ void wait_async0() {
#if __has_builtin(__builtin_amdgcn_s_wait_asynccnt)
  __builtin_amdgcn_s_wait_asynccnt(0);
#else
  asm volatile("s_wait_asynccnt 0x0" ::: "memory");
#endif
}

// ---- WMMA fragment helpers (wave32, layouts per CDNA5 ISA 7.12.2) ----------

// A-matrix 16x32 bf16: lane row = lane&15; VGPR j (j<4): K=2j+8h, (j>=4): K=16+2(j-4)+8h
__device__ __forceinline__ v16bf load_a_frag(const __bf16* rowp, int lane) {
  const int h8 = (lane >> 4) * 8;
  v16bf f;
#pragma unroll
  for (int j = 0; j < 8; ++j) {
    const int k = ((j < 4) ? (2 * j) : (16 + 2 * (j - 4))) + h8;
    f[2 * j]     = rowp[k];
    f[2 * j + 1] = rowp[k + 1];
  }
  return f;
}

// B-matrix 32x16 bf16: lane col = lane&15; VGPR j (j<4): K=2j+8h, (j>=4): K=16+2(j-4)+8h
__device__ __forceinline__ v16bf load_b_frag(const __bf16* rowp, int lane) {
  const int h8 = (lane >> 4) * 8;
  v16bf f;
#pragma unroll
  for (int j = 0; j < 8; ++j) {
    const int k = ((j < 4) ? (2 * j) : (16 + 2 * (j - 4))) + h8;
    f[2 * j]     = rowp[k];
    f[2 * j + 1] = rowp[k + 1];
  }
  return f;
}

__device__ __forceinline__ v8f wmma_bf16(v16bf a, v16bf b, v8f c) {
  return __builtin_amdgcn_wmma_f32_16x16x32_bf16(false, a, false, b, (short)0, c,
                                                 false, false);
}

// ---- fp32 -> bf16 conversion ----------------------------------------------

__global__ __launch_bounds__(256) void cvt_f32_bf16(const float* __restrict__ in,
                                                    __bf16* __restrict__ out, int n) {
  int i = blockIdx.x * 256 + threadIdx.x;
  if (i < n) out[i] = (__bf16)in[i];
}

// ---- Generic WMMA GEMM: C[M,N] = A[M,K] @ B[K,N] + bias -------------------
// Tile 128x128, BK=32 (one WMMA K-step per iter), 8 waves of 64x32.
// A: async-to-LDS double buffered; B: register-pipelined transpose staging.

__global__ __launch_bounds__(256) void gemm_wmma_bf16(
    const __bf16* __restrict__ A, const __bf16* __restrict__ B,
    const float* __restrict__ bias, void* __restrict__ C,
    int M, int N, int K, int out_f32) {
  __shared__ __bf16 As[2][128][36];  // [buf][m][k]
  __shared__ __bf16 Bs[128][36];     // [n][k] (transposed)

  const int tid  = threadIdx.x;
  const int wave = tid >> 5, lane = tid & 31;
  const int wm = wave >> 2, wn = wave & 3;  // 2x4 wave grid, wavetile 64x32
  const int bm = blockIdx.y * 128, bn = blockIdx.x * 128;
  const int mrow = lane & 15, hh = lane >> 4;

  const int a_r = tid >> 3, a_c = (tid & 7) * 4;     // A chunks: rows a_r+32i
  const int b_k = tid >> 5, b_c = (tid & 31) * 4;    // B chunks: k-rows b_k+8i

  auto issue_a = [&](int k0, int buf) {
#pragma unroll
    for (int i = 0; i < 4; ++i) {
      const int r = a_r + i * 32;
      async_b64(&As[buf][r][a_c], A + (size_t)(bm + r) * K + k0 + a_c);
    }
  };
  auto load_b = [&](int k0, uint2* breg) {
#pragma unroll
    for (int i = 0; i < 4; ++i)
      breg[i] = *(const uint2*)(B + (size_t)(k0 + b_k + i * 8) * N + bn + b_c);
  };
  auto store_b = [&](const uint2* breg) {
#pragma unroll
    for (int i = 0; i < 4; ++i) {
      const __bf16* pv = (const __bf16*)&breg[i];
      const int kr = b_k + i * 8;
      Bs[b_c + 0][kr] = pv[0];
      Bs[b_c + 1][kr] = pv[1];
      Bs[b_c + 2][kr] = pv[2];
      Bs[b_c + 3][kr] = pv[3];
    }
  };

  uint2 breg[4];
  issue_a(0, 0);
  load_b(0, breg);

  const v8f vzero = {};
  v8f acc[4][2];
#pragma unroll
  for (int mt = 0; mt < 4; ++mt)
#pragma unroll
    for (int nt = 0; nt < 2; ++nt) acc[mt][nt] = vzero;

  int cur = 0;
  for (int k0 = 0; k0 < K; k0 += 32) {
    wait_async0();          // this wave's As[cur] chunks are in LDS
    store_b(breg);          // Bs for this tile
    __syncthreads();        // all waves' As[cur]/Bs visible

    const bool next = (k0 + 32) < K;
    if (next) {
      issue_a(k0 + 32, cur ^ 1);  // async loads overlap compute below
      load_b(k0 + 32, breg);      // global loads overlap compute below
      if (k0 + 64 < K)            // global_prefetch_b8 of the B tile after next
        __builtin_prefetch(B + (size_t)(k0 + 64 + b_k) * N + bn + b_c, 0, 0);
    }

    v16bf af[4], bfr[2];
#pragma unroll
    for (int mt = 0; mt < 4; ++mt)
      af[mt] = load_a_frag(&As[cur][wm * 64 + mt * 16 + mrow][0], lane);
#pragma unroll
    for (int nt = 0; nt < 2; ++nt)
      bfr[nt] = load_b_frag(&Bs[wn * 32 + nt * 16 + mrow][0], lane);
#pragma unroll
    for (int mt = 0; mt < 4; ++mt)
#pragma unroll
      for (int nt = 0; nt < 2; ++nt)
        acc[mt][nt] = wmma_bf16(af[mt], bfr[nt], acc[mt][nt]);

    __syncthreads();        // protect As[cur]/Bs before next overwrite
    cur ^= 1;
  }

#pragma unroll
  for (int mt = 0; mt < 4; ++mt)
#pragma unroll
    for (int nt = 0; nt < 2; ++nt) {
      const int n = bn + wn * 32 + nt * 16 + mrow;
      const float bv = bias[n];
#pragma unroll
      for (int r = 0; r < 8; ++r) {
        const int m = bm + wm * 64 + mt * 16 + r + 8 * hh;
        const float v = acc[mt][nt][r] + bv;
        if (out_f32) ((float*)C)[(size_t)m * N + n] = v;
        else         ((__bf16*)C)[(size_t)m * N + n] = (__bf16)v;
      }
    }
}

// ---- RoPE: x_r = x_row @ W (128x64); x[0:32]=rot, x[32:64]=0 ---------------

__global__ __launch_bounds__(32) void rope_proj(__bf16* __restrict__ X,
                                                const float* __restrict__ W) {
  __shared__ float row[HD];
  const int s = blockIdx.x, h = blockIdx.y, t = threadIdx.x;
  __bf16* xp = X + (size_t)s * HIDN + h * HD;
#pragma unroll
  for (int i = 0; i < 4; ++i) row[t + i * 32] = (float)xp[t + i * 32];
  __syncthreads();

  float a = 0.f, b = 0.f;
#pragma unroll 8
  for (int d = 0; d < HD; ++d) {
    const float x = row[d];
    a += x * W[d * 64 + t];
    b += x * W[d * 64 + t + 32];
  }
  const float inv_freq = __powf(10000.0f, -(float)(2 * t) / 64.0f);
  float sn, cs;
  __sincosf((float)s * inv_freq, &sn, &cs);
  xp[t]      = (__bf16)(a * cs - b * sn);
  xp[t + 32] = (__bf16)0.0f;
}

// ---- Flash attention: softmax(QK^T/sqrt(D)) V, all matmuls via WMMA --------
// grid (S/128, H); 8 waves x 16 q rows; 64-key tiles.
// K tile: async-to-LDS straight copy; V tile: register-pipelined transpose.

__global__ __launch_bounds__(256) void mla_flash_attn(
    const __bf16* __restrict__ Q, const __bf16* __restrict__ Kg,
    const __bf16* __restrict__ Vg, __bf16* __restrict__ ctx) {
  __shared__ __bf16 Ks[64][132];    // [key][d]
  __shared__ __bf16 Vt[128][68];    // [d][key]
  __shared__ __bf16 Ps[8][16][72];  // per-wave P staging, 16B-aligned rows

  const int tid = threadIdx.x, wave = tid >> 5, lane = tid & 31;
  const int h = blockIdx.y;
  const int mrow = lane & 15, hh = lane >> 4;
  const int qrow = blockIdx.x * 128 + wave * 16;
  const float scale = 0.08838834764831845f;  // 1/sqrt(128)

  const int k_r = tid >> 5, k_c = (tid & 31) * 4;  // tile chunks: rows k_r+8i

  auto issue_k = [&](int kt) {
#pragma unroll
    for (int i = 0; i < 8; ++i) {
      const int r = k_r + i * 8;
      async_b64(&Ks[r][k_c], Kg + (size_t)(kt + r) * HIDN + h * HD + k_c);
    }
  };
  auto load_v = [&](int kt, uint2* vreg) {
#pragma unroll
    for (int i = 0; i < 8; ++i)
      vreg[i] = *(const uint2*)(Vg + (size_t)(kt + k_r + i * 8) * HIDN + h * HD + k_c);
  };
  auto store_v = [&](const uint2* vreg) {
#pragma unroll
    for (int i = 0; i < 8; ++i) {
      const __bf16* pv = (const __bf16*)&vreg[i];
      const int r = k_r + i * 8;
      Vt[k_c + 0][r] = pv[0];
      Vt[k_c + 1][r] = pv[1];
      Vt[k_c + 2][r] = pv[2];
      Vt[k_c + 3][r] = pv[3];
    }
  };

  // Q fragments (A-layout), loaded once, reused over all key tiles
  v16bf qf[4];
  const __bf16* qbase = Q + (size_t)(qrow + mrow) * HIDN + h * HD;
#pragma unroll
  for (int kk = 0; kk < 4; ++kk) qf[kk] = load_a_frag(qbase + kk * 32, lane);

  const v8f vzero = {};
  v8f o[8];
#pragma unroll
  for (int dt = 0; dt < 8; ++dt) o[dt] = vzero;
  float mi[8], li[8];
#pragma unroll
  for (int r = 0; r < 8; ++r) { mi[r] = -1e30f; li[r] = 0.f; }

  uint2 vreg[8];
  load_v(0, vreg);

  for (int kt = 0; kt < S_LEN; kt += 64) {
    issue_k(kt);        // async K copy, overlaps the V transpose stores below
    store_v(vreg);
    wait_async0();
    __syncthreads();

    const bool next = (kt + 64) < S_LEN;
    if (next) {
      load_v(kt + 64, vreg);  // next V tile loads overlap compute
      if (tid < 128)          // warm L2/L0 for the next K tile
        __builtin_prefetch(
            Kg + (size_t)(kt + 64 + (tid >> 1)) * HIDN + h * HD + (tid & 1) * 64, 0, 0);
    }

    // S = Q K^T : M=16 q rows, N=64 keys, K=128 d
    v8f s[4];
#pragma unroll
    for (int nt = 0; nt < 4; ++nt) {
      s[nt] = vzero;
#pragma unroll
      for (int kk = 0; kk < 4; ++kk) {
        v16bf b = load_b_frag(&Ks[nt * 16 + mrow][kk * 32], lane);
        s[nt] = wmma_bf16(qf[kk], b, s[nt]);
      }
    }

    // online softmax (C-layout rows: r + 8*hh, cols across lanes 0..15)
    float mnew[8], al[8], rs[8];
#pragma unroll
    for (int r = 0; r < 8; ++r) {
      float mx = -1e30f;
#pragma unroll
      for (int nt = 0; nt < 4; ++nt) mx = fmaxf(mx, s[nt][r]);
      mx *= scale;
#pragma unroll
      for (int d2 = 1; d2 < 16; d2 <<= 1) mx = fmaxf(mx, __shfl_xor(mx, d2, 32));
      mnew[r] = fmaxf(mi[r], mx);
      al[r] = __expf(mi[r] - mnew[r]);
      rs[r] = 0.f;
    }
#pragma unroll
    for (int nt = 0; nt < 4; ++nt)
#pragma unroll
      for (int r = 0; r < 8; ++r) {
        const float p = __expf(s[nt][r] * scale - mnew[r]);
        s[nt][r] = p;
        rs[r] += p;
      }
#pragma unroll
    for (int r = 0; r < 8; ++r) {
#pragma unroll
      for (int d2 = 1; d2 < 16; d2 <<= 1) rs[r] += __shfl_xor(rs[r], d2, 32);
      li[r] = li[r] * al[r] + rs[r];
      mi[r] = mnew[r];
    }
#pragma unroll
    for (int dt = 0; dt < 8; ++dt)
#pragma unroll
      for (int r = 0; r < 8; ++r) o[dt][r] *= al[r];

    // P (C-layout) -> per-wave LDS -> A-layout (DS in-order within a wave)
#pragma unroll
    for (int nt = 0; nt < 4; ++nt)
#pragma unroll
      for (int r = 0; r < 8; ++r)
        Ps[wave][r + 8 * hh][nt * 16 + mrow] = (__bf16)s[nt][r];

    // O += P V : M=16, N=128 d, K=64 keys
#pragma unroll
    for (int kk2 = 0; kk2 < 2; ++kk2) {
      v16bf a = load_a_frag(&Ps[wave][mrow][kk2 * 32], lane);
#pragma unroll
      for (int dt = 0; dt < 8; ++dt) {
        v16bf b = load_b_frag(&Vt[dt * 16 + mrow][kk2 * 32], lane);
        o[dt] = wmma_bf16(a, b, o[dt]);
      }
    }
    __syncthreads();   // protect Ks/Vt before next tile's staging
  }

#pragma unroll
  for (int dt = 0; dt < 8; ++dt)
#pragma unroll
    for (int r = 0; r < 8; ++r) {
      const float v = o[dt][r] / li[r];
      ctx[(size_t)(qrow + r + 8 * hh) * HIDN + h * HD + dt * 16 + mrow] = (__bf16)v;
    }
}

// ---------------------------------------------------------------------------

extern "C" void kernel_launch(void* const* d_in, const int* in_sizes, int n_in,
                              void* d_out, int out_size, void* d_ws, size_t ws_size,
                              hipStream_t stream) {
  (void)in_sizes; (void)n_in; (void)out_size; (void)ws_size;
  const float* hs   = (const float*)d_in[0];
  const float* wdkv = (const float*)d_in[1];
  const float* bdkv = (const float*)d_in[2];
  const float* wuk  = (const float*)d_in[3];
  const float* buk  = (const float*)d_in[4];
  const float* wuv  = (const float*)d_in[5];
  const float* buv  = (const float*)d_in[6];
  const float* wdq  = (const float*)d_in[7];
  const float* bdq  = (const float*)d_in[8];
  const float* wuq  = (const float*)d_in[9];
  const float* buq  = (const float*)d_in[10];
  const float* wqr  = (const float*)d_in[11];
  const float* wkr  = (const float*)d_in[12];
  const float* wo   = (const float*)d_in[13];
  const float* bo   = (const float*)d_in[14];
  float* out = (float*)d_out;

  __bf16* ws = (__bf16*)d_ws;
  size_t off = 0;
  auto alloc = [&](size_t n) { __bf16* p = ws + off; off += n; return p; };
  __bf16* hsb   = alloc((size_t)S_LEN * HIDN);
  __bf16* wdkvb = alloc((size_t)HIDN * 512);
  __bf16* wukb  = alloc((size_t)512 * HIDN);
  __bf16* wuvb  = alloc((size_t)512 * HIDN);
  __bf16* wdqb  = alloc((size_t)HIDN * 1536);
  __bf16* wuqb  = alloc((size_t)1536 * HIDN);
  __bf16* wob   = alloc((size_t)HIDN * HIDN);
  __bf16* ckv   = alloc((size_t)S_LEN * 512);
  __bf16* cq    = alloc((size_t)S_LEN * 1536);
  __bf16* kb    = alloc((size_t)S_LEN * HIDN);
  __bf16* vb    = alloc((size_t)S_LEN * HIDN);
  __bf16* qb    = alloc((size_t)S_LEN * HIDN);
  __bf16* ctxb  = alloc((size_t)S_LEN * HIDN);

  auto cvt = [&](const float* src, __bf16* dst, int n) {
    cvt_f32_bf16<<<(n + 255) / 256, 256, 0, stream>>>(src, dst, n);
  };
  cvt(hs,   hsb,   S_LEN * HIDN);
  cvt(wdkv, wdkvb, HIDN * 512);
  cvt(wuk,  wukb,  512 * HIDN);
  cvt(wuv,  wuvb,  512 * HIDN);
  cvt(wdq,  wdqb,  HIDN * 1536);
  cvt(wuq,  wuqb,  1536 * HIDN);
  cvt(wo,   wob,   HIDN * HIDN);

  auto gemm = [&](const __bf16* A, const __bf16* B, const float* bias, void* C,
                  int M, int N, int K, int out_f32) {
    gemm_wmma_bf16<<<dim3(N / 128, M / 128), 256, 0, stream>>>(A, B, bias, C, M, N, K,
                                                               out_f32);
  };

  gemm(hsb, wdkvb, bdkv, ckv, S_LEN, 512, HIDN, 0);   // c_kv
  gemm(ckv, wukb,  buk,  kb,  S_LEN, HIDN, 512, 0);   // k
  gemm(ckv, wuvb,  buv,  vb,  S_LEN, HIDN, 512, 0);   // v
  gemm(hsb, wdqb,  bdq,  cq,  S_LEN, 1536, HIDN, 0);  // c_q
  gemm(cq,  wuqb,  buq,  qb,  S_LEN, HIDN, 1536, 0);  // q

  rope_proj<<<dim3(S_LEN, NH), 32, 0, stream>>>(qb, wqr);
  rope_proj<<<dim3(S_LEN, NH), 32, 0, stream>>>(kb, wkr);

  mla_flash_attn<<<dim3(S_LEN / 128, NH), 256, 0, stream>>>(qb, kb, vb, ctxb);

  gemm(ctxb, wob, bo, out, S_LEN, HIDN, HIDN, 1);     // output (f32)
}
